// EncoderBlock_3968549781692
// MI455X (gfx1250) — matrix operand order repaired and checked by reading, not compiled, not run on hardware
//
#include <hip/hip_runtime.h>
#include <math.h>

typedef __attribute__((ext_vector_type(16))) _Float16 v16h;
typedef __attribute__((ext_vector_type(8)))  _Float16 v8h;
typedef __attribute__((ext_vector_type(8)))  float    v8f;

#define B_   64
#define C_   128
#define L_   400
#define H_   8
#define DK_  16
#define KW_  7
#define LP_  416              // L padded to multiple of 32 for attention K-dim
#define NT_  (L_ / 16)        // 25 column tiles
#define CL_  (C_ * L_)        // 51200
#define BCL_ ((size_t)B_ * C_ * L_)

__device__ __forceinline__ v16h cat8(v8h lo, v8h hi) {
  v16h r;
#pragma unroll
  for (int j = 0; j < 8; ++j) { r[j] = lo[j]; r[j + 8] = hi[j]; }
  return r;
}

// ---------------------------------------------------------------------------
// Positional encoding table: pos[c*L + l] = sin(l * freq(c) + phase(c))
// ---------------------------------------------------------------------------
__global__ void pos_kernel(float* __restrict__ pos) {
  int i = blockIdx.x * 256 + threadIdx.x;
  if (i >= CL_) return;
  int c = i / L_, l = i % L_;
  float f, ph;
  if ((c & 1) == 0) { f = powf(10000.f, -(float)c / (float)C_); ph = 0.f; }
  else              { f = -powf(10000.f, (1.f - (float)c) / (float)C_); ph = 1.57079632679f; }
  pos[i] = sinf((float)l * f + ph);
}

__global__ void add_pos(const float* __restrict__ x, const float* __restrict__ pos,
                        float* __restrict__ res) {
  size_t i = (size_t)blockIdx.x * 256 + threadIdx.x;
  if (i >= BCL_) return;
  res[i] = x[i] + pos[i % CL_];
}

__global__ void cvt_f16(const float* __restrict__ src, _Float16* __restrict__ dst, int n) {
  int i = blockIdx.x * 256 + threadIdx.x;
  if (i < n) dst[i] = (_Float16)src[i];
}

// ---------------------------------------------------------------------------
// Per-batch LayerNorm stats: mean/var over the whole (C,L) slab.
// ---------------------------------------------------------------------------
__global__ __launch_bounds__(256)
void ln_reduce(const float* __restrict__ X, float* __restrict__ mu, float* __restrict__ rstd) {
  __shared__ float ss[256], sq[256];
  const int b = blockIdx.x;
  const float* xb = X + (size_t)b * CL_;
  float s = 0.f, q = 0.f;
  for (int i = threadIdx.x; i < CL_; i += 256) { float v = xb[i]; s += v; q += v * v; }
  ss[threadIdx.x] = s; sq[threadIdx.x] = q;
  __syncthreads();
  for (int st = 128; st > 0; st >>= 1) {
    if (threadIdx.x < st) { ss[threadIdx.x] += ss[threadIdx.x + st]; sq[threadIdx.x] += sq[threadIdx.x + st]; }
    __syncthreads();
  }
  if (threadIdx.x == 0) {
    float m = ss[0] / (float)CL_;
    float v = sq[0] / (float)CL_ - m * m;
    mu[b] = m;
    rstd[b] = rsqrtf(v + 1e-5f);
  }
}

// ---------------------------------------------------------------------------
// LN apply + pack: writes f32 xn[b][c][l] (for dwconv) AND the f16 GEMM
// B-operand tiled layout xt[((b*NT + l/16)*16 + l%16)*128 + c] via an LDS
// 16x16 transpose (both global accesses coalesced).
// grid = (NT, C/16, B), block = 256.
// ---------------------------------------------------------------------------
__global__ __launch_bounds__(256)
void ln_apply_pack(const float* __restrict__ X, const float* __restrict__ mu,
                   const float* __restrict__ rstd, const float* __restrict__ g,
                   const float* __restrict__ beta, float* __restrict__ xn,
                   _Float16* __restrict__ xt) {
  __shared__ _Float16 lds[16][17];
  const int l0 = blockIdx.x * 16;
  const int c0 = blockIdx.y * 16;
  const int b  = blockIdx.z;
  const int i = threadIdx.x >> 4;   // c within tile
  const int j = threadIdx.x & 15;   // l within tile
  const float m = mu[b], rs = rstd[b];

  const size_t cl = (size_t)(c0 + i) * L_ + l0 + j;
  float z = (X[(size_t)b * CL_ + cl] - m) * rs * g[cl] + beta[cl];
  xn[(size_t)b * CL_ + cl] = z;
  lds[i][j] = (_Float16)z;
  __syncthreads();

  const int ii = threadIdx.x & 15;  // c within tile (now fastest)
  const int jj = threadIdx.x >> 4;  // l within tile
  xt[(((size_t)b * NT_ + blockIdx.x) * 16 + jj) * 128 + c0 + ii] = lds[ii][jj];
}

// Same packing for an f32 [C][L] activation (dwconv output, attention output).
__global__ __launch_bounds__(256)
void pack_xt(const float* __restrict__ X, _Float16* __restrict__ xt) {
  __shared__ _Float16 lds[16][17];
  const int l0 = blockIdx.x * 16;
  const int c0 = blockIdx.y * 16;
  const int b  = blockIdx.z;
  const int i = threadIdx.x >> 4;
  const int j = threadIdx.x & 15;
  lds[i][j] = (_Float16)X[(size_t)b * CL_ + (size_t)(c0 + i) * L_ + l0 + j];
  __syncthreads();
  const int ii = threadIdx.x & 15;
  const int jj = threadIdx.x >> 4;
  xt[(((size_t)b * NT_ + blockIdx.x) * 16 + jj) * 128 + c0 + ii] = lds[ii][jj];
}

// ---------------------------------------------------------------------------
// Depthwise conv, K=7, same padding (cross-correlation as in lax.conv).
// ---------------------------------------------------------------------------
__global__ __launch_bounds__(128)
void dwconv(const float* __restrict__ X, const float* __restrict__ W,
            const float* __restrict__ bias, float* __restrict__ Y) {
  const int c = blockIdx.x, b = blockIdx.y;
  const float* xb = X + ((size_t)b * C_ + c) * L_;
  float* yb = Y + ((size_t)b * C_ + c) * L_;
  float w[KW_];
#pragma unroll
  for (int k = 0; k < KW_; ++k) w[k] = W[c * KW_ + k];
  const float bb = bias[c];
  for (int l = threadIdx.x; l < L_; l += 128) {
    float s = bb;
#pragma unroll
    for (int k = 0; k < KW_; ++k) {
      int src = l + k - KW_ / 2;
      if (src >= 0 && src < L_) s += w[k] * xb[src];
    }
    yb[l] = s;
  }
}

// ---------------------------------------------------------------------------
// Per-batch GEMM: Z(128x400) = W(128x128,f16) @ X(128x400,f16 tiled)
// grid = (NT, B), block = 256 (8 waves); wave w owns M-tile w, block owns one
// N-tile. The X tile (4KB, already in B-operand layout in global memory) is
// staged into LDS with one global_load_async_to_lds_b128 per lane (ASYNCcnt).
// Epilogue modes:
//   0: out = relu(Z + bias) + Res           (f32, pointwise conv / FC)
//   1: out = Z + bias + Res                 (f32, attention O projection)
//   2: outH[((b*H+h)*LP + l)*32 + d] = (Z + bias) * scale   (Q / K, padded f16)
//   3: outH[((b*H+h)*DK + d)*LP + l] = Z + bias             (V transposed f16)
// ---------------------------------------------------------------------------
__global__ __launch_bounds__(256)
void gemm_cl(const _Float16* __restrict__ W, const float* __restrict__ bias,
             const _Float16* __restrict__ Xt, const float* __restrict__ Res,
             float* __restrict__ outF, _Float16* __restrict__ outH,
             float scale, int mode) {
  __shared__ __align__(32) _Float16 ldsX[16 * 128];
  const int b  = blockIdx.y;
  const int l0 = blockIdx.x * 16;
  const int t  = threadIdx.x;
  const int lane = t & 31;
  const int wave = t >> 5;

  // Async-copy the 4KB X tile (layout [n][k], f16) straight into LDS:
  // 256 lanes x 16B contiguous chunks.
  {
    const _Float16* src = Xt + (((size_t)b * NT_ + blockIdx.x) * 16) * 128 + t * 8;
    unsigned ldsAddr = (unsigned)(size_t)(&ldsX[t * 8]);
    asm volatile("global_load_async_to_lds_b128 %0, %1, off"
                 :: "v"(ldsAddr), "v"(src) : "memory");
    asm volatile("s_wait_asynccnt 0" ::: "memory");
  }
  __syncthreads();

  const int n    = lane & 15;
  const int hi   = lane >> 4;            // half-wave select
  const int row0 = wave * 16 + n;        // A-operand row for this lane
  v8f acc = {};
#pragma unroll
  for (int kk = 0; kk < 128; kk += 32) {
    const _Float16* wr = W + (size_t)row0 * 128 + kk + hi * 16;
    v16h a = *(const v16h*)wr;
    v16h bf = *(const v16h*)(&ldsX[n * 128 + kk + hi * 16]);
    acc = __builtin_amdgcn_wmma_f32_16x16x32_f16(false, a, false, bf,
                                                 (short)0, acc, false, false);
  }

  const int lg = l0 + n;
  const int mbase = wave * 16 + hi * 8;  // C/D layout: vgpr r -> row mbase+r
  if (mode <= 1) {
    float* outb = outF + (size_t)b * CL_;
    const float* resb = Res + (size_t)b * CL_;
#pragma unroll
    for (int r = 0; r < 8; ++r) {
      const int row = mbase + r;
      float z = acc[r] + bias[row];
      if (mode == 0) z = fmaxf(z, 0.f);
      z += resb[(size_t)row * L_ + lg];
      outb[(size_t)row * L_ + lg] = z;
    }
  } else if (mode == 2) {
#pragma unroll
    for (int r = 0; r < 8; ++r) {
      const int row = mbase + r;
      const int h = row >> 4, d = row & 15;
      float z = (acc[r] + bias[row]) * scale;
      outH[(((size_t)b * H_ + h) * LP_ + lg) * 32 + d] = (_Float16)z;
    }
  } else {
#pragma unroll
    for (int r = 0; r < 8; ++r) {
      const int row = mbase + r;
      const int h = row >> 4, d = row & 15;
      outH[(((size_t)b * H_ + h) * DK_ + d) * LP_ + lg] = (_Float16)(acc[r] + bias[row]);
    }
  }
}

// ---------------------------------------------------------------------------
// Flash attention: one wave per (b, h, 16-query tile). Q prescaled by 1/sqrt(C).
// Q rows padded to K=32 with zeros; keys padded to 416 and masked to -1e30.
// ---------------------------------------------------------------------------
__global__ __launch_bounds__(32)
void flash_attn(const _Float16* __restrict__ Q, const _Float16* __restrict__ Km,
                const _Float16* __restrict__ Vt, const float* __restrict__ mask,
                float* __restrict__ attn) {
  __shared__ __align__(32) _Float16 ldsP[16 * 32];
  const int l0 = blockIdx.x * 16;
  const int h  = blockIdx.y;
  const int b  = blockIdx.z;
  const int lane = threadIdx.x;
  const int n  = lane & 15;
  const int hi = lane >> 4;
  const size_t bh = (size_t)b * H_ + h;

  // Q A-fragment (16x32, d=16..31 are zero pad)
  const _Float16* qr = Q + (bh * LP_ + l0 + n) * 32 + hi * 16;
  const v16h aq = *(const v16h*)qr;

  float mi[8], li[8];
  v8f o = {};
#pragma unroll
  for (int r = 0; r < 8; ++r) { mi[r] = -3.0e38f; li[r] = 0.f; }

  for (int jt = 0; jt < LP_; jt += 32) {
    // S1 = Q @ K[jt..jt+15]^T ; S2 = Q @ K[jt+16..jt+31]^T
    const _Float16* kr1 = Km + (bh * LP_ + jt + n) * 32 + hi * 16;
    const _Float16* kr2 = Km + (bh * LP_ + jt + 16 + n) * 32 + hi * 16;
    v8f s1 = {}, s2 = {};
    s1 = __builtin_amdgcn_wmma_f32_16x16x32_f16(false, aq, false, *(const v16h*)kr1,
                                                (short)0, s1, false, false);
    s2 = __builtin_amdgcn_wmma_f32_16x16x32_f16(false, aq, false, *(const v16h*)kr2,
                                                (short)0, s2, false, false);

    const int key1 = jt + n, key2 = jt + 16 + n;
    const float m1 = (key1 < L_) ? mask[b * L_ + key1] : 0.f;
    const float m2 = (key2 < L_) ? mask[b * L_ + key2] : 0.f;

#pragma unroll
    for (int r = 0; r < 8; ++r) {
      float s1v = s1[r] * m1 - (1.f - m1) * 1e30f;
      float s2v = s2[r] * m2 - (1.f - m2) * 1e30f;
      // row max across the 16 lanes of this half-wave
      float tmax = fmaxf(s1v, s2v);
#pragma unroll
      for (int off = 1; off < 16; off <<= 1)
        tmax = fmaxf(tmax, __shfl_xor(tmax, off, 32));
      const float mnew  = fmaxf(mi[r], tmax);
      const float alpha = expf(mi[r] - mnew);
      const float p1 = expf(s1v - mnew);
      const float p2 = expf(s2v - mnew);
      float psum = p1 + p2;
#pragma unroll
      for (int off = 1; off < 16; off <<= 1)
        psum += __shfl_xor(psum, off, 32);
      li[r] = li[r] * alpha + psum;
      mi[r] = mnew;
      o[r] *= alpha;
      // stash P (C-layout -> row-major tile) for A-operand re-load
      ldsP[(r + hi * 8) * 32 + n]      = (_Float16)p1;
      ldsP[(r + hi * 8) * 32 + 16 + n] = (_Float16)p2;
    }
    __syncthreads();

    // P as A-operand (16x32)
    const _Float16* pr = &ldsP[n * 32 + hi * 16];
    const v16h ap = *(const v16h*)pr;
    // V as B-operand: Vt[d][jt + k], contiguous in k
    const _Float16* vr = Vt + (bh * DK_ + n) * LP_ + jt + hi * 16;
    o = __builtin_amdgcn_wmma_f32_16x16x32_f16(false, ap, false, *(const v16h*)vr,
                                               (short)0, o, false, false);
    __syncthreads();
  }

#pragma unroll
  for (int r = 0; r < 8; ++r) {
    const int m = r + hi * 8;
    attn[(size_t)b * CL_ + (size_t)(h * DK_ + n) * L_ + l0 + m] = o[r] / li[r];
  }
}

// ---------------------------------------------------------------------------
extern "C" void kernel_launch(void* const* d_in, const int* in_sizes, int n_in,
                              void* d_out, int out_size, void* d_ws, size_t ws_size,
                              hipStream_t stream) {
  (void)in_sizes; (void)n_in; (void)out_size; (void)ws_size;
  const float* x    = (const float*)d_in[0];
  const float* mask = (const float*)d_in[1];
  const float* dw_w = (const float*)d_in[2];
  const float* dw_b = (const float*)d_in[3];
  const float* pw_w = (const float*)d_in[4];
  const float* pw_b = (const float*)d_in[5];
  const float* wq = (const float*)d_in[6];  const float* bq = (const float*)d_in[7];
  const float* wk = (const float*)d_in[8];  const float* bk = (const float*)d_in[9];
  const float* wv = (const float*)d_in[10]; const float* bv = (const float*)d_in[11];
  const float* wo = (const float*)d_in[12]; const float* bo = (const float*)d_in[13];
  const float* fcw = (const float*)d_in[14]; const float* fcb = (const float*)d_in[15];
  const float* lng = (const float*)d_in[16]; const float* lnb = (const float*)d_in[17];
  float* out = (float*)d_out;

  char* base = (char*)d_ws;
  size_t off = 0;
  auto take = [&](size_t bytes) -> char* {
    char* p = base + off;
    off = (off + bytes + 255) & ~(size_t)255;
    return p;
  };
  float*    pos  = (float*)take(CL_ * 4);
  float*    res  = (float*)take(BCL_ * 4);
  float*    xn   = (float*)take(BCL_ * 4);
  float*    y    = (float*)take(BCL_ * 4);       // conv temp, reused as attn out
  float*    mu   = (float*)take(B_ * 4);
  float*    rstd = (float*)take(B_ * 4);
  _Float16* xt   = (_Float16*)take(BCL_ * 2);    // tiled f16 GEMM input
  _Float16* wh   = (_Float16*)take((size_t)9 * C_ * C_ * 2);  // pw0..3,q,k,v,o,fc
  const size_t qkBytes = (size_t)B_ * H_ * LP_ * 32 * 2;
  const size_t vtBytes = (size_t)B_ * H_ * DK_ * LP_ * 2;
  _Float16* qb = (_Float16*)take(qkBytes);
  _Float16* kb = (_Float16*)take(qkBytes);
  _Float16* vt = (_Float16*)take(vtBytes);

  hipMemsetAsync(qb, 0, qkBytes, stream);
  hipMemsetAsync(kb, 0, qkBytes, stream);
  hipMemsetAsync(vt, 0, vtBytes, stream);

  pos_kernel<<<(CL_ + 255) / 256, 256, 0, stream>>>(pos);

  const float* wsrc[9] = {pw_w + 0 * C_ * C_, pw_w + 1 * C_ * C_,
                          pw_w + 2 * C_ * C_, pw_w + 3 * C_ * C_,
                          wq, wk, wv, wo, fcw};
  for (int i = 0; i < 9; ++i)
    cvt_f16<<<(C_ * C_ + 255) / 256, 256, 0, stream>>>(wsrc[i], wh + (size_t)i * C_ * C_, C_ * C_);

  const int nb = (int)((BCL_ + 255) / 256);
  const dim3 pgrid(NT_, C_ / 16, B_);
  add_pos<<<nb, 256, 0, stream>>>(x, pos, res);
  ln_reduce<<<B_, 256, 0, stream>>>(res, mu, rstd);
  ln_apply_pack<<<pgrid, 256, 0, stream>>>(res, mu, rstd, lng, lnb, xn, xt);

  const dim3 ggrid(NT_, B_);
  for (int i = 0; i < 4; ++i) {
    dwconv<<<dim3(C_, B_), 128, 0, stream>>>(xn, dw_w + (size_t)i * C_ * KW_,
                                             dw_b + (size_t)i * C_, y);
    pack_xt<<<pgrid, 256, 0, stream>>>(y, xt);
    gemm_cl<<<ggrid, 256, 0, stream>>>(wh + (size_t)i * C_ * C_, pw_b + (size_t)i * C_,
                                       xt, res, res, (_Float16*)nullptr, 1.f, 0);
    ln_reduce<<<B_, 256, 0, stream>>>(res, mu, rstd);
    ln_apply_pack<<<pgrid, 256, 0, stream>>>(res, mu, rstd, lng, lnb, xn, xt);
  }

  const float qscale = 0.08838834764831845f;  // 1/sqrt(C)
  gemm_cl<<<ggrid, 256, 0, stream>>>(wh + (size_t)4 * C_ * C_, bq, xt, nullptr, nullptr, qb, qscale, 2);
  gemm_cl<<<ggrid, 256, 0, stream>>>(wh + (size_t)5 * C_ * C_, bk, xt, nullptr, nullptr, kb, 1.f, 2);
  gemm_cl<<<ggrid, 256, 0, stream>>>(wh + (size_t)6 * C_ * C_, bv, xt, nullptr, nullptr, vt, 1.f, 3);

  flash_attn<<<dim3(NT_, H_, B_), 32, 0, stream>>>(qb, kb, vt, mask, y);

  pack_xt<<<pgrid, 256, 0, stream>>>(y, xt);
  gemm_cl<<<ggrid, 256, 0, stream>>>(wh + (size_t)7 * C_ * C_, bo, xt, res, res, (_Float16*)nullptr, 1.f, 1);
  ln_reduce<<<B_, 256, 0, stream>>>(res, mu, rstd);
  ln_apply_pack<<<pgrid, 256, 0, stream>>>(res, mu, rstd, lng, lnb, xn, xt);
  gemm_cl<<<ggrid, 256, 0, stream>>>(wh + (size_t)8 * C_ * C_, fcb, xt, res, out, (_Float16*)nullptr, 1.f, 0);
}
